// Transformer_19782619366019
// MI455X (gfx1250) — compile-verified, matrix operand order checked by reading, not run on hardware
//
#include <hip/hip_runtime.h>
#include <hip/hip_bf16.h>
#include <math.h>

typedef __bf16 bf16_t;
typedef __attribute__((ext_vector_type(16))) __bf16 v16bf;
typedef __attribute__((ext_vector_type(8)))  float  v8f;

#define S_LEN 2048
#define D_DIM 1024
#define F_DIM 4096
#define H_NUM 16
#define DH_DIM 64

union FragU { v16bf v; uint4 u[2]; };

// Load a 16x32 bf16 fragment (A-layout; B uses identical addressing because we
// store B transposed). Lane l: row = l&15, K elements (l>>4)*8 + {0..7} into
// elems 0..7 and +16 into elems 8..15 — matches CDNA5 16-bit A/B VGPR striping.
__device__ __forceinline__ v16bf load_frag(const bf16_t* __restrict__ base, int ld) {
  int lane = threadIdx.x & 31;
  const bf16_t* p = base + (size_t)(lane & 15) * ld + ((lane >> 4) << 3);
  FragU r;
  r.u[0] = *(const uint4*)(p);
  r.u[1] = *(const uint4*)(p + 16);
  return r.v;
}

__device__ __forceinline__ v8f wmma_bf16(v16bf a, v16bf b, v8f c) {
  return __builtin_amdgcn_wmma_f32_16x16x32_bf16(false, a, false, b, (short)0, c, false, false);
}

// ---------------------------------------------------------------- converts
__global__ void cvt_f32_bf16_kernel(const float* __restrict__ in, bf16_t* __restrict__ out, size_t n) {
  size_t i = (size_t)blockIdx.x * blockDim.x + threadIdx.x;
  if (i < n) out[i] = (bf16_t)in[i];
}

// Wt[n*K + k] = (bf16) W[k*N + n]   (W is KxN row-major, Wt is NxK row-major)
__global__ void transpose_cvt_kernel(const float* __restrict__ W, bf16_t* __restrict__ Wt, int K, int N) {
  size_t i = (size_t)blockIdx.x * blockDim.x + threadIdx.x;
  size_t total = (size_t)K * N;
  if (i >= total) return;
  int k = (int)(i % (size_t)K);
  int n = (int)(i / (size_t)K);
  Wt[i] = (bf16_t)W[(size_t)k * N + n];
}

// At[c*R + r] = A[r*C + c]   (bf16 -> bf16)
__global__ void transpose_bf16_kernel(const bf16_t* __restrict__ A, bf16_t* __restrict__ At, int R, int C) {
  size_t i = (size_t)blockIdx.x * blockDim.x + threadIdx.x;
  size_t total = (size_t)R * C;
  if (i >= total) return;
  int r = (int)(i % (size_t)R);
  int c = (int)(i / (size_t)R);
  At[i] = A[(size_t)r * C + c];
}

// ---------------------------------------------------------------- GEMM
// C[M,N] = A[M,K] @ Bt[N,K]^T + bias, bf16 inputs, fp32 accumulate.
// mode 0: store bf16;  mode 1: exact GELU then store bf16;  mode 2: store f32.
// Block = 256 threads (8 waves = 4(M) x 2(N)). Wave tile 32x64, block 128x128.
// K-loop double-buffered so fragment loads overlap the 8 WMMAs per step.
__global__ __launch_bounds__(256) void gemm_bf16_kernel(
    const bf16_t* __restrict__ A, const bf16_t* __restrict__ Bt,
    const float* __restrict__ bias, void* __restrict__ Cout,
    int M, int N, int K, int mode) {
  int wave = threadIdx.x >> 5;
  int lane = threadIdx.x & 31;
  int wm = wave & 3;
  int wn = wave >> 2;
  int m0 = blockIdx.y * 128 + wm * 32;
  int n0 = blockIdx.x * 128 + wn * 64;

  const bf16_t* Ap0 = A + (size_t)m0 * K;
  const bf16_t* Ap1 = A + (size_t)(m0 + 16) * K;
  const bf16_t* Bp0 = Bt + (size_t)(n0 +  0) * K;
  const bf16_t* Bp1 = Bt + (size_t)(n0 + 16) * K;
  const bf16_t* Bp2 = Bt + (size_t)(n0 + 32) * K;
  const bf16_t* Bp3 = Bt + (size_t)(n0 + 48) * K;

  v8f acc[2][4];
  #pragma unroll
  for (int mi = 0; mi < 2; ++mi)
    #pragma unroll
    for (int f = 0; f < 4; ++f)
      #pragma unroll
      for (int i = 0; i < 8; ++i) acc[mi][f][i] = 0.0f;

  // prologue: fragments for k0 = 0
  v16bf a0 = load_frag(Ap0, K);
  v16bf a1 = load_frag(Ap1, K);
  v16bf b0 = load_frag(Bp0, K);
  v16bf b1 = load_frag(Bp1, K);
  v16bf b2 = load_frag(Bp2, K);
  v16bf b3 = load_frag(Bp3, K);

  for (int k0 = 32; k0 < K; k0 += 32) {
    // prefetch further ahead (lowers to global_prefetch_b8; speculative OOB ok)
    __builtin_prefetch(Ap0 + k0 + 96, 0, 1);
    __builtin_prefetch(Bp0 + k0 + 96, 0, 1);
    // next-step fragments into fresh registers (overlaps with WMMAs below)
    v16bf na0 = load_frag(Ap0 + k0, K);
    v16bf na1 = load_frag(Ap1 + k0, K);
    v16bf nb0 = load_frag(Bp0 + k0, K);
    v16bf nb1 = load_frag(Bp1 + k0, K);
    v16bf nb2 = load_frag(Bp2 + k0, K);
    v16bf nb3 = load_frag(Bp3 + k0, K);

    acc[0][0] = wmma_bf16(a0, b0, acc[0][0]);
    acc[1][0] = wmma_bf16(a1, b0, acc[1][0]);
    acc[0][1] = wmma_bf16(a0, b1, acc[0][1]);
    acc[1][1] = wmma_bf16(a1, b1, acc[1][1]);
    acc[0][2] = wmma_bf16(a0, b2, acc[0][2]);
    acc[1][2] = wmma_bf16(a1, b2, acc[1][2]);
    acc[0][3] = wmma_bf16(a0, b3, acc[0][3]);
    acc[1][3] = wmma_bf16(a1, b3, acc[1][3]);

    a0 = na0; a1 = na1; b0 = nb0; b1 = nb1; b2 = nb2; b3 = nb3;
  }
  // epilogue k-step
  acc[0][0] = wmma_bf16(a0, b0, acc[0][0]);
  acc[1][0] = wmma_bf16(a1, b0, acc[1][0]);
  acc[0][1] = wmma_bf16(a0, b1, acc[0][1]);
  acc[1][1] = wmma_bf16(a1, b1, acc[1][1]);
  acc[0][2] = wmma_bf16(a0, b2, acc[0][2]);
  acc[1][2] = wmma_bf16(a1, b2, acc[1][2]);
  acc[0][3] = wmma_bf16(a0, b3, acc[0][3]);
  acc[1][3] = wmma_bf16(a1, b3, acc[1][3]);

  int half = lane >> 4;
  int nlo  = lane & 15;
  #pragma unroll
  for (int mi = 0; mi < 2; ++mi) {
    #pragma unroll
    for (int f = 0; f < 4; ++f) {
      int n = n0 + f * 16 + nlo;
      float bv = bias[n];
      #pragma unroll
      for (int r = 0; r < 8; ++r) {
        int m = m0 + mi * 16 + r + half * 8;
        float v = acc[mi][f][r] + bv;
        if (mode == 1) v = 0.5f * v * (1.0f + erff(v * 0.70710678118654752f));
        if (mode == 2) ((float*)Cout)[(size_t)m * N + n] = v;
        else           ((bf16_t*)Cout)[(size_t)m * N + n] = (bf16_t)v;
      }
    }
  }
}

// ---------------------------------------------------------------- attention
// Flash-style. Grid (S/128, H), block 256 (8 waves). Wave handles one 16-row
// q tile of one head. Q,K: bf16 [S, D] (head cols h*64..). Vt: bf16 [D, S].
// K-frags for the next step and V-frags for this step are loaded before the
// softmax so global traffic overlaps the exp/shuffle chain.
__global__ __launch_bounds__(256) void attention_kernel(
    const bf16_t* __restrict__ Q, const bf16_t* __restrict__ Kc,
    const bf16_t* __restrict__ Vt, const int* __restrict__ amask,
    bf16_t* __restrict__ Out) {
  __shared__ __align__(16) bf16_t ldsP[8][16 * 32];
  int wave = threadIdx.x >> 5;
  int lane = threadIdx.x & 31;
  int h  = blockIdx.y;
  int q0 = blockIdx.x * 128 + wave * 16;
  int half = lane >> 4;
  int nlo  = lane & 15;

  v16bf qf0 = load_frag(Q + (size_t)q0 * D_DIM + h * DH_DIM +  0, D_DIM);
  v16bf qf1 = load_frag(Q + (size_t)q0 * D_DIM + h * DH_DIM + 32, D_DIM);

  v8f acc[4];
  float Mrow[8], Lrow[8];
  #pragma unroll
  for (int f = 0; f < 4; ++f)
    #pragma unroll
    for (int i = 0; i < 8; ++i) acc[f][i] = 0.0f;
  #pragma unroll
  for (int r = 0; r < 8; ++r) { Mrow[r] = -1e30f; Lrow[r] = 0.0f; }

  // prologue K fragments (sk0 = 0): kf[j][kchunk]
  v16bf kf00 = load_frag(Kc + (size_t)0  * D_DIM + h * DH_DIM +  0, D_DIM);
  v16bf kf01 = load_frag(Kc + (size_t)0  * D_DIM + h * DH_DIM + 32, D_DIM);
  v16bf kf10 = load_frag(Kc + (size_t)16 * D_DIM + h * DH_DIM +  0, D_DIM);
  v16bf kf11 = load_frag(Kc + (size_t)16 * D_DIM + h * DH_DIM + 32, D_DIM);

  for (int sk0 = 0; sk0 < S_LEN; sk0 += 32) {
    v8f sc[2];
    #pragma unroll
    for (int j = 0; j < 2; ++j)
      #pragma unroll
      for (int i = 0; i < 8; ++i) sc[j][i] = 0.0f;

    sc[0] = wmma_bf16(qf0, kf00, sc[0]);
    sc[0] = wmma_bf16(qf1, kf01, sc[0]);
    sc[1] = wmma_bf16(qf0, kf10, sc[1]);
    sc[1] = wmma_bf16(qf1, kf11, sc[1]);

    // next-step K fragments (wrap on last iter -> harmless redundant loads)
    int skn = (sk0 + 32) & (S_LEN - 1);
    v16bf nk00 = load_frag(Kc + (size_t)(skn +  0) * D_DIM + h * DH_DIM +  0, D_DIM);
    v16bf nk01 = load_frag(Kc + (size_t)(skn +  0) * D_DIM + h * DH_DIM + 32, D_DIM);
    v16bf nk10 = load_frag(Kc + (size_t)(skn + 16) * D_DIM + h * DH_DIM +  0, D_DIM);
    v16bf nk11 = load_frag(Kc + (size_t)(skn + 16) * D_DIM + h * DH_DIM + 32, D_DIM);
    // this-step V fragments, loaded early to overlap the softmax
    v16bf vf[4];
    #pragma unroll
    for (int f = 0; f < 4; ++f)
      vf[f] = load_frag(Vt + (size_t)(h * DH_DIM + f * 16) * S_LEN + sk0, S_LEN);

    int am0 = amask[sk0 + nlo];
    int am1 = amask[sk0 + 16 + nlo];
    #pragma unroll
    for (int r = 0; r < 8; ++r) {
      float s0 = sc[0][r] * 0.125f;  // 1/sqrt(64)
      float s1 = sc[1][r] * 0.125f;
      sc[0][r] = am0 ? s0 : -1e9f;
      sc[1][r] = am1 ? s1 : -1e9f;
    }

    // online softmax; row r+8*half lives in the 16 lanes of this half
    #pragma unroll
    for (int r = 0; r < 8; ++r) {
      float t = fmaxf(sc[0][r], sc[1][r]);
      #pragma unroll
      for (int off = 1; off < 16; off <<= 1)
        t = fmaxf(t, __shfl_xor(t, off, 32));
      float mnew  = fmaxf(Mrow[r], t);
      float scale = __expf(Mrow[r] - mnew);
      Mrow[r] = mnew;
      float p0 = __expf(sc[0][r] - mnew);
      float p1 = __expf(sc[1][r] - mnew);
      sc[0][r] = p0; sc[1][r] = p1;
      float rs = p0 + p1;
      #pragma unroll
      for (int off = 1; off < 16; off <<= 1)
        rs += __shfl_xor(rs, off, 32);
      Lrow[r] = Lrow[r] * scale + rs;
      acc[0][r] *= scale; acc[1][r] *= scale;
      acc[2][r] *= scale; acc[3][r] *= scale;
    }

    // repack P (C-layout) -> A-layout via wave-private LDS (DS ops in-order per wave)
    bf16_t* lp = ldsP[wave];
    #pragma unroll
    for (int r = 0; r < 8; ++r) {
      int m = r + half * 8;
      lp[m * 32 + nlo]      = (bf16_t)sc[0][r];
      lp[m * 32 + 16 + nlo] = (bf16_t)sc[1][r];
    }
    v16bf pf = load_frag(lp, 32);

    #pragma unroll
    for (int f = 0; f < 4; ++f)
      acc[f] = wmma_bf16(pf, vf[f], acc[f]);

    kf00 = nk00; kf01 = nk01; kf10 = nk10; kf11 = nk11;
  }

  #pragma unroll
  for (int r = 0; r < 8; ++r) {
    float inv = 1.0f / fmaxf(Lrow[r], 1e-20f);
    int m = q0 + r + half * 8;
    #pragma unroll
    for (int f = 0; f < 4; ++f)
      Out[(size_t)m * D_DIM + h * DH_DIM + f * 16 + nlo] = (bf16_t)(acc[f][r] * inv);
  }
}

// ---------------------------------------------------------------- layernorm
// out = gamma * (x - mean)/sqrt(var + 1e-12) + beta, x = inp + resid.
__global__ __launch_bounds__(256) void layernorm_kernel(
    const float* __restrict__ inp, const float* __restrict__ resid,
    const float* __restrict__ gamma, const float* __restrict__ beta,
    float* __restrict__ outf, bf16_t* __restrict__ outb) {
  __shared__ float red[256];
  int row = blockIdx.x;
  int tid = threadIdx.x;
  const int PER = D_DIM / 256;  // 4
  float xs[PER];
  float lsum = 0.0f;
  #pragma unroll
  for (int i = 0; i < PER; ++i) {
    int c = tid + i * 256;
    xs[i] = inp[(size_t)row * D_DIM + c] + resid[(size_t)row * D_DIM + c];
    lsum += xs[i];
  }
  red[tid] = lsum; __syncthreads();
  for (int s = 128; s > 0; s >>= 1) { if (tid < s) red[tid] += red[tid + s]; __syncthreads(); }
  float mean = red[0] * (1.0f / D_DIM);
  __syncthreads();
  float lvar = 0.0f;
  #pragma unroll
  for (int i = 0; i < PER; ++i) { float d = xs[i] - mean; lvar += d * d; }
  red[tid] = lvar; __syncthreads();
  for (int s = 128; s > 0; s >>= 1) { if (tid < s) red[tid] += red[tid + s]; __syncthreads(); }
  float inv = rsqrtf(red[0] * (1.0f / D_DIM) + 1e-12f);
  #pragma unroll
  for (int i = 0; i < PER; ++i) {
    int c = tid + i * 256;
    float v = gamma[c] * ((xs[i] - mean) * inv) + beta[c];
    outf[(size_t)row * D_DIM + c] = v;
    if (outb) outb[(size_t)row * D_DIM + c] = (bf16_t)v;
  }
}

// ---------------------------------------------------------------- launch
extern "C" void kernel_launch(void* const* d_in, const int* in_sizes, int n_in,
                              void* d_out, int out_size, void* d_ws, size_t ws_size,
                              hipStream_t stream) {
  const float* x    = (const float*)d_in[0];
  const int*   mask = (const int*)  d_in[1];
  const float* Wq = (const float*)d_in[2];  const float* bq = (const float*)d_in[3];
  const float* Wk = (const float*)d_in[4];  const float* bk = (const float*)d_in[5];
  const float* Wv = (const float*)d_in[6];  const float* bv = (const float*)d_in[7];
  const float* Wo = (const float*)d_in[8];  const float* bo = (const float*)d_in[9];
  const float* g1 = (const float*)d_in[10]; const float* be1 = (const float*)d_in[11];
  const float* W1 = (const float*)d_in[12]; const float* c1 = (const float*)d_in[13];
  const float* W2 = (const float*)d_in[14]; const float* c2 = (const float*)d_in[15];
  const float* g2 = (const float*)d_in[16]; const float* be2 = (const float*)d_in[17];
  float* out = (float*)d_out;

  char* ws = (char*)d_ws;
  size_t off = 0;
  auto alloc = [&](size_t bytes) -> void* {
    void* p = ws + off;
    off = (off + bytes + 255) & ~(size_t)255;
    return p;
  };
  const size_t SD = (size_t)S_LEN * D_DIM, SF = (size_t)S_LEN * F_DIM;
  const size_t DD = (size_t)D_DIM * D_DIM, DF = (size_t)D_DIM * F_DIM;

  bf16_t* X16   = (bf16_t*)alloc(SD * 2);
  bf16_t* Wqt   = (bf16_t*)alloc(DD * 2);
  bf16_t* Wkt   = (bf16_t*)alloc(DD * 2);
  bf16_t* Wvt   = (bf16_t*)alloc(DD * 2);
  bf16_t* Wot   = (bf16_t*)alloc(DD * 2);
  bf16_t* W1t   = (bf16_t*)alloc(DF * 2);
  bf16_t* W2t   = (bf16_t*)alloc(DF * 2);
  bf16_t* Q16   = (bf16_t*)alloc(SD * 2);
  bf16_t* K16   = (bf16_t*)alloc(SD * 2);
  bf16_t* V16   = (bf16_t*)alloc(SD * 2);
  bf16_t* Vt16  = (bf16_t*)alloc(SD * 2);
  bf16_t* Att16 = (bf16_t*)alloc(SD * 2);
  float*  Of32  = (float*) alloc(SD * 4);
  float*  A1f   = (float*) alloc(SD * 4);
  bf16_t* A1b   = (bf16_t*)alloc(SD * 2);
  bf16_t* H16   = (bf16_t*)alloc(SF * 2);
  float*  Mf32  = (float*) alloc(SD * 4);
  (void)ws_size; (void)in_sizes; (void)n_in; (void)out_size;

  // 1. converts / transposed weights
  cvt_f32_bf16_kernel<<<(SD + 255) / 256, 256, 0, stream>>>(x, X16, SD);
  transpose_cvt_kernel<<<(DD + 255) / 256, 256, 0, stream>>>(Wq, Wqt, D_DIM, D_DIM);
  transpose_cvt_kernel<<<(DD + 255) / 256, 256, 0, stream>>>(Wk, Wkt, D_DIM, D_DIM);
  transpose_cvt_kernel<<<(DD + 255) / 256, 256, 0, stream>>>(Wv, Wvt, D_DIM, D_DIM);
  transpose_cvt_kernel<<<(DD + 255) / 256, 256, 0, stream>>>(Wo, Wot, D_DIM, D_DIM);
  transpose_cvt_kernel<<<(DF + 255) / 256, 256, 0, stream>>>(W1, W1t, D_DIM, F_DIM); // W1t: [F, D]
  transpose_cvt_kernel<<<(DF + 255) / 256, 256, 0, stream>>>(W2, W2t, F_DIM, D_DIM); // W2t: [D, F]

  // 2. Q/K/V projections (bf16 out)
  dim3 gP(D_DIM / 128, S_LEN / 128);
  gemm_bf16_kernel<<<gP, 256, 0, stream>>>(X16, Wqt, bq, Q16, S_LEN, D_DIM, D_DIM, 0);
  gemm_bf16_kernel<<<gP, 256, 0, stream>>>(X16, Wkt, bk, K16, S_LEN, D_DIM, D_DIM, 0);
  gemm_bf16_kernel<<<gP, 256, 0, stream>>>(X16, Wvt, bv, V16, S_LEN, D_DIM, D_DIM, 0);

  // 3. Vt[d][s] = V[s][d]  -> [H][DH][S]
  transpose_bf16_kernel<<<(SD + 255) / 256, 256, 0, stream>>>(V16, Vt16, S_LEN, D_DIM);

  // 4. attention
  dim3 gA(S_LEN / 128, H_NUM);
  attention_kernel<<<gA, 256, 0, stream>>>(Q16, K16, Vt16, mask, Att16);

  // 5. output projection (f32 out) + LN1 (residual x)
  gemm_bf16_kernel<<<gP, 256, 0, stream>>>(Att16, Wot, bo, Of32, S_LEN, D_DIM, D_DIM, 2);
  layernorm_kernel<<<S_LEN, 256, 0, stream>>>(Of32, x, g1, be1, A1f, A1b);

  // 6. FFN
  dim3 gF1(F_DIM / 128, S_LEN / 128);
  gemm_bf16_kernel<<<gF1, 256, 0, stream>>>(A1b, W1t, c1, H16, S_LEN, F_DIM, D_DIM, 1); // GELU
  gemm_bf16_kernel<<<gP, 256, 0, stream>>>(H16, W2t, c2, Mf32, S_LEN, D_DIM, F_DIM, 2);

  // 7. LN2 (residual a1) -> d_out
  layernorm_kernel<<<S_LEN, 256, 0, stream>>>(Mf32, A1f, g2, be2, out, (bf16_t*)nullptr);
}
